// RNNForMPC_77807627534357
// MI455X (gfx1250) — matrix operand order repaired and checked by reading, not compiled, not run on hardware
//
#include <hip/hip_runtime.h>

// ---------------- problem constants ----------------
#define BB   64
#define TT   512
#define IN_  128
#define HH   512
#define GG   (4 * HH)       // 2048
#define OUTN 64
#define BT   (BB * TT)      // 32768

typedef unsigned short u16;
typedef __attribute__((ext_vector_type(16))) __bf16 bf16x16;
typedef __attribute__((ext_vector_type(8)))  float  floatx8;
typedef __attribute__((ext_vector_type(4)))  unsigned uintx4;
typedef __attribute__((ext_vector_type(8)))  int      intx8;
typedef __attribute__((ext_vector_type(4)))  int      intx4;

// ---------------- helpers ----------------
__device__ __forceinline__ u16 f2bf(float f) {
    unsigned u = __float_as_uint(f);
    unsigned r = u + 0x7FFFu + ((u >> 16) & 1u);
    return (u16)(r >> 16);
}

__device__ __forceinline__ float sigf(float x) {
    return 1.0f / (1.0f + __expf(-x));
}

union FragU {
    bf16x16 v;
    u16     h[16];
    uint4   q[2];
};

// A fragment: 16x32 bf16, source [m][k] with row stride ldk, tile = &A[k0]
// lanes 0-15: row m=lane, K {0..7, 16..23}; lanes 16-31: row m=lane-16, K {8..15, 24..31}
__device__ __forceinline__ bf16x16 load_a_frag(const u16* tile, size_t ldk, int lane) {
    int m  = lane & 15;
    int hi = (lane >> 4) << 3;   // 0 or 8
    FragU u;
    u.q[0] = *(const uint4*)(tile + (size_t)m * ldk + hi);
    u.q[1] = *(const uint4*)(tile + (size_t)m * ldk + 16 + hi);
    return u.v;
}

// B fragment: 32x16 bf16; B stored as [n][k] (row n = output column), tile = &B[n0*ldk + k0]
__device__ __forceinline__ bf16x16 load_b_frag(const u16* tile, size_t ldk, int lane) {
    int n    = lane & 15;
    int koff = (lane >> 4) << 3; // 0 or 8
    FragU u;
    const u16* p = tile + (size_t)n * ldk + koff;
    u.q[0] = *(const uint4*)p;
    u.q[1] = *(const uint4*)(p + 16);
    return u.v;
}

// ---------------- Tensor Data Mover: contiguous block (global -> LDS) ----------------
// 1-D tile: n_dwords 4-byte elements, per CDNA5 D# bitfields (CDNA5 ISA).
__device__ __forceinline__ void tdm_load_to_lds(unsigned lds_off, const void* gsrc,
                                                unsigned n_dwords) {
    unsigned long long ga = (unsigned long long)gsrc;
    uintx4 g0;
    g0.x = 1u;                                   // count=1 valid descriptor
    g0.y = lds_off;                              // lds_addr (bytes)
    g0.z = (unsigned)ga;                         // global_addr[31:0]
    g0.w = (unsigned)((ga >> 32) & 0x01FFFFFFu) | (2u << 30);  // addr[56:32] | type=2
    intx8 g1;
    g1[0] = (int)(2u << 16);                     // data_size = 2 (4 bytes)
    g1[1] = (int)((n_dwords & 0xFFFFu) << 16);   // tensor_dim0[15:0]  @ bits[63:48]
    g1[2] = (int)(((n_dwords >> 16) & 0xFFFFu)   // tensor_dim0[31:16] @ bits[79:64]
                  | (1u << 16));                 // tensor_dim1 = 1    @ bits[111:80]
    g1[3] = (int)((n_dwords & 0xFFFFu) << 16);   // tile_dim0          @ bits[127:112]
    g1[4] = 1;                                   // tile_dim1 = 1      @ bits[143:128]
    g1[5] = (int)n_dwords;                       // tensor_dim0_stride @ bits[207:160]
    g1[6] = 0;
    g1[7] = 0;
    intx4 gz = (intx4)0;
#if defined(__clang_major__) && (__clang_major__ >= 23)
    intx8 gz8 = (intx8)0;
    __builtin_amdgcn_tensor_load_to_lds(g0, g1, gz, gz, gz8, 0);
#else
    __builtin_amdgcn_tensor_load_to_lds(g0, g1, gz, gz, 0);
#endif
}

// ---------------- prep kernels ----------------
__global__ void cvt_bf16_kernel(const float* __restrict__ src, u16* __restrict__ dst, int n) {
    int i = blockIdx.x * blockDim.x + threadIdx.x;
    int stride = gridDim.x * blockDim.x;
    for (; i < n; i += stride) dst[i] = f2bf(src[i]);
}

__global__ void add_bias_kernel(const float* __restrict__ a, const float* __restrict__ b,
                                float* __restrict__ o, int n) {
    int i = blockIdx.x * blockDim.x + threadIdx.x;
    if (i < n) o[i] = a[i] + b[i];
}

__global__ void init_bar_kernel(unsigned* bar) {
    if (threadIdx.x < 64) bar[threadIdx.x] = 0u;
}

// ---------------- generic WMMA GEMM:  C = A[M,K](bf16) * B[N,K]^T(bf16) + bias[N] ----------------
// Each wave computes one 16(M) x 16(N) tile.
// fragC=0: A rows contiguous (stride K), C row-major [M][N]. blockIdx.x = M/16 tiles.
//   (aRowStride must equal K)
// fragC=1: blockIdx.x = t*(BB/16) + bTile; A row b at fixed t lives at A + b*aRowStride + t*K;
//   C written in WMMA fragment layout: C[((blockIdx.x)*(N/16) + n0/16)*256 + lane*8 + r].
__global__ __launch_bounds__(256) void wmma_gemm_bias_kernel(
    const u16* __restrict__ A, const u16* __restrict__ Bm,
    const float* __restrict__ bias, float* __restrict__ C,
    int N, int K, size_t aRowStride, int fragC)
{
    const int lane  = threadIdx.x & 31;
    const int wave  = threadIdx.x >> 5;
    const int waves = blockDim.x >> 5;
    const int n0    = (blockIdx.y * waves + wave) * 16;
    if (n0 >= N) return;

    int t = 0, bTile = blockIdx.x;
    if (fragC) { t = blockIdx.x >> 2; bTile = blockIdx.x & 3; }   // BB/16 == 4

    floatx8 acc = (floatx8)0.0f;
    const u16* Abase = A + (size_t)bTile * 16 * aRowStride + (size_t)t * K;
    const u16* Bbase = Bm + (size_t)n0 * K;
#pragma unroll 4
    for (int k0 = 0; k0 < K; k0 += 32) {
        bf16x16 af = load_a_frag(Abase + k0, aRowStride, lane);
        bf16x16 bf = load_b_frag(Bbase + k0, K, lane);
        acc = __builtin_amdgcn_wmma_f32_16x16x32_bf16(false, af, false, bf, (short)0, acc,
                                                      false, false);
    }

    const int   n  = n0 + (lane & 15);
    const float bv = bias[n];
    if (fragC) {
        float* cp = C + (((size_t)blockIdx.x) * (N >> 4) + (n0 >> 4)) * 256 + lane * 8;
#pragma unroll
        for (int r = 0; r < 8; ++r) cp[r] = acc[r] + bv;
    } else {
        const int mbase = bTile * 16 + ((lane >> 4) << 3);
#pragma unroll
        for (int r = 0; r < 8; ++r) C[(size_t)(mbase + r) * N + n] = acc[r] + bv;
    }
}

// ---------------- persistent LSTM recurrence ----------------
// grid = 64 WGs (4 row-groups x 16 col-groups), 256 threads (8 waves).
// Per step: g = xg[t] + h_{t-1} @ Whh^T (WMMA bf16 -> f32).
// Whh fragments live in REGISTERS for all 512 steps (16 frags = 128 VGPRs/lane).
// h_{t-1} slice DMA'd to LDS by the Tensor Data Mover each step; xg arrives in
// fragment-native layout (one b128-pair load per lane); gates via LDS; c in registers;
// h broadcast via global hbuf + monotonic atomic grid barrier.

__device__ __forceinline__ void grid_sync(unsigned* bar, unsigned target) {
    __syncthreads();
    if (threadIdx.x == 0) {
        __builtin_amdgcn_fence(__ATOMIC_RELEASE, "agent");
        __hip_atomic_fetch_add(bar, 1u, __ATOMIC_RELAXED, __HIP_MEMORY_SCOPE_AGENT);
        while (__hip_atomic_load(bar, __ATOMIC_RELAXED, __HIP_MEMORY_SCOPE_AGENT) < target) {
            __builtin_amdgcn_s_sleep(1);
        }
    }
    __syncthreads();
    __builtin_amdgcn_fence(__ATOMIC_ACQUIRE, "agent");
}

__global__ __launch_bounds__(256, 1) void lstm_rec_kernel(
    const float* __restrict__ xgf,   // fragment layout [t][bTile][nTile][lane][8]
    const u16*   __restrict__ whh,   // [4H][H] bf16
    const float* __restrict__ h0l,   // [B][H] fp32
    const float* __restrict__ c0l,   // [B][H] fp32
    u16* __restrict__ hbuf,          // [B][H] bf16 shared state
    u16* __restrict__ y,             // [B][T][H] bf16 layer output
    float* __restrict__ hn, float* __restrict__ cn,  // [B][H] fp32 finals
    unsigned* __restrict__ bar, int nblocks)
{
    __shared__ u16   a_lds[16 * HH];     // 16KB: h_{t-1} rows [m0, m0+16)
    __shared__ float glds[16 * 136];     // gate tile

    const int tid  = threadIdx.x;
    const int lane = tid & 31;
    const int wave = tid >> 5;
    const int m0   = (blockIdx.x >> 4) * 16;   // batch rows
    const int c0c  = (blockIdx.x & 15) * 32;   // hidden cols

    // two elementwise cells per thread
    const int cell0 = tid, cell1 = tid + 256;
    const int row0 = cell0 >> 5, j0 = cell0 & 31;
    const int row1 = cell1 >> 5, j1 = cell1 & 31;
    const int b0 = m0 + row0, b1 = m0 + row1;
    const int hc0 = c0c + j0, hc1 = c0c + j1;

    float c_0 = c0l[b0 * HH + hc0];
    float c_1 = c0l[b1 * HH + hc1];
    hbuf[b0 * HH + hc0] = f2bf(h0l[b0 * HH + hc0]);
    hbuf[b1 * HH + hc1] = f2bf(h0l[b1 * HH + hc1]);

    // wave -> gate subtile: gate s = wave/2, within-gate col offset jw
    const int s     = wave >> 1;
    const int jw    = (wave & 1) * 16;
    const int nbase = s * HH + c0c + jw;      // global gate column base
    const int nn    = lane & 15;
    const int mhi   = (lane >> 4) << 3;

    // ---- preload the ENTIRE Whh strip for this wave into registers (16 frags) ----
    bf16x16 bfr[16];
    {
        const u16* Bbase = whh + (size_t)nbase * HH;
#pragma unroll
        for (int kk = 0; kk < 16; ++kk)
            bfr[kk] = load_b_frag(Bbase + kk * 32, HH, lane);
    }

    const unsigned a_lds_off = (unsigned)(size_t)(void*)a_lds;   // LDS byte offset
    const u16*     hsrc      = hbuf + (size_t)m0 * HH;           // 16KB contiguous slice

    // per-wave fragment pointer into xgf for this (m0, nbase) tile
    const float* xfrag = xgf + (((size_t)(m0 >> 4)) * (GG >> 4) + (nbase >> 4)) * 256 + lane * 8;
    const size_t xstep = (size_t)4 * (GG >> 4) * 256;   // advance one timestep

    unsigned tick = 0;
    grid_sync(bar, ++tick * nblocks);

    for (int t = 0; t < TT; ++t) {
        // ---- TDM: DMA h_{t-1}[m0:m0+16, :] (16KB) into LDS (wave 0 drives) ----
        if (wave == 0) {
            tdm_load_to_lds(a_lds_off, hsrc, (16 * HH * 2) / 4);
            __builtin_amdgcn_s_wait_tensorcnt(0);
        }

        // ---- accumulator = precomputed input gates, fragment-native (1 contiguous 32B/lane) ----
        const float* xp = xfrag + (size_t)t * xstep;
        floatx8 acc = *(const floatx8*)xp;
        if (t + 1 < TT) __builtin_prefetch(xp + xstep, 0, 1);

        __syncthreads();   // a_lds ready

        // ---- g += h_{t-1} @ Whh^T : A from LDS, B from registers ----
#pragma unroll
        for (int kk = 0; kk < 16; ++kk) {
            bf16x16 af = load_a_frag(a_lds + kk * 32, HH, lane);
            acc = __builtin_amdgcn_wmma_f32_16x16x32_bf16(false, af, false, bfr[kk], (short)0,
                                                          acc, false, false);
        }

        // ---- spill gate tile to LDS: WG-local gate col = wave*16 + lane%16 = s*32 + j ----
        {
            const int col = wave * 16 + nn;
#pragma unroll
            for (int r = 0; r < 8; ++r) glds[(mhi + r) * 136 + col] = acc[r];
        }
        __syncthreads();

        // ---- fused gate nonlinearities, 2 cells/thread, c lives in registers ----
        {
            float i_ = glds[row0 * 136 + j0];
            float f_ = glds[row0 * 136 + 32 + j0];
            float g_ = glds[row0 * 136 + 64 + j0];
            float o_ = glds[row0 * 136 + 96 + j0];
            c_0 = sigf(f_) * c_0 + sigf(i_) * tanhf(g_);
            float h = sigf(o_) * tanhf(c_0);
            u16 hb = f2bf(h);
            hbuf[b0 * HH + hc0] = hb;
            y[((size_t)b0 * TT + t) * HH + hc0] = hb;
            if (t == TT - 1) { hn[b0 * HH + hc0] = h; cn[b0 * HH + hc0] = c_0; }
        }
        {
            float i_ = glds[row1 * 136 + j1];
            float f_ = glds[row1 * 136 + 32 + j1];
            float g_ = glds[row1 * 136 + 64 + j1];
            float o_ = glds[row1 * 136 + 96 + j1];
            c_1 = sigf(f_) * c_1 + sigf(i_) * tanhf(g_);
            float h = sigf(o_) * tanhf(c_1);
            u16 hb = f2bf(h);
            hbuf[b1 * HH + hc1] = hb;
            y[((size_t)b1 * TT + t) * HH + hc1] = hb;
            if (t == TT - 1) { hn[b1 * HH + hc1] = h; cn[b1 * HH + hc1] = c_1; }
        }
        grid_sync(bar, ++tick * nblocks);
    }
}

// ---------------- host launch ----------------
static inline size_t align256(size_t x) { return (x + 255) & ~(size_t)255; }

extern "C" void kernel_launch(void* const* d_in, const int* in_sizes, int n_in,
                              void* d_out, int out_size, void* d_ws, size_t ws_size,
                              hipStream_t stream) {
    const float* x    = (const float*)d_in[0];
    const float* h0   = (const float*)d_in[1];
    const float* c0   = (const float*)d_in[2];
    const float* Wih0 = (const float*)d_in[3];
    const float* Whh0 = (const float*)d_in[4];
    const float* bih0 = (const float*)d_in[5];
    const float* bhh0 = (const float*)d_in[6];
    const float* Wih1 = (const float*)d_in[7];
    const float* Whh1 = (const float*)d_in[8];
    const float* bih1 = (const float*)d_in[9];
    const float* bhh1 = (const float*)d_in[10];
    const float* Wfc  = (const float*)d_in[11];
    const float* bfc  = (const float*)d_in[12];

    float* outf = (float*)d_out;
    float* hn   = outf + (size_t)BT * OUTN;              // [2][B][H]
    float* cn   = hn + 2 * BB * HH;                      // [2][B][H]

    // workspace carve-out
    char*  ws  = (char*)d_ws;
    size_t off = 0;
    unsigned* bar = (unsigned*)(ws + off); off = align256(off + 64 * sizeof(unsigned));
    u16* xbf   = (u16*)(ws + off); off = align256(off + (size_t)BT * IN_ * 2);
    u16* wih0p = (u16*)(ws + off); off = align256(off + (size_t)GG * IN_ * 2);
    u16* whh0p = (u16*)(ws + off); off = align256(off + (size_t)GG * HH * 2);
    u16* wih1p = (u16*)(ws + off); off = align256(off + (size_t)GG * HH * 2);
    u16* whh1p = (u16*)(ws + off); off = align256(off + (size_t)GG * HH * 2);
    u16* wfcp  = (u16*)(ws + off); off = align256(off + (size_t)OUTN * HH * 2);
    float* bsum0 = (float*)(ws + off); off = align256(off + (size_t)GG * 4);
    float* bsum1 = (float*)(ws + off); off = align256(off + (size_t)GG * 4);
    u16* hbuf  = (u16*)(ws + off); off = align256(off + (size_t)BB * HH * 2);
    u16* y0    = (u16*)(ws + off); off = align256(off + (size_t)BT * HH * 2);
    u16* y1    = (u16*)(ws + off); off = align256(off + (size_t)BT * HH * 2);
    float* xg  = (float*)(ws + off); off = align256(off + (size_t)BT * GG * 4);
    (void)ws_size; (void)n_in; (void)in_sizes; (void)out_size;

    // --- prep: zero barriers, cast to bf16, fold biases ---
    init_bar_kernel<<<1, 64, 0, stream>>>(bar);
    cvt_bf16_kernel<<<2048, 256, 0, stream>>>(x,    xbf,   BT * IN_);
    cvt_bf16_kernel<<<512,  256, 0, stream>>>(Wih0, wih0p, GG * IN_);
    cvt_bf16_kernel<<<1024, 256, 0, stream>>>(Whh0, whh0p, GG * HH);
    cvt_bf16_kernel<<<1024, 256, 0, stream>>>(Wih1, wih1p, GG * HH);
    cvt_bf16_kernel<<<1024, 256, 0, stream>>>(Whh1, whh1p, GG * HH);
    cvt_bf16_kernel<<<128,  256, 0, stream>>>(Wfc,  wfcp,  OUTN * HH);
    add_bias_kernel<<<8, 256, 0, stream>>>(bih0, bhh0, bsum0, GG);
    add_bias_kernel<<<8, 256, 0, stream>>>(bih1, bhh1, bsum1, GG);

    // --- layer 0: input GEMM into fragment-layout xg, then persistent recurrence ---
    wmma_gemm_bias_kernel<<<dim3(TT * (BB / 16), GG / 128), 256, 0, stream>>>(
        xbf, wih0p, bsum0, xg, GG, IN_, /*aRowStride=*/(size_t)TT * IN_, /*fragC=*/1);
    lstm_rec_kernel<<<64, 256, 0, stream>>>(
        xg, whh0p, h0 + 0 * BB * HH, c0 + 0 * BB * HH,
        hbuf, y0, hn + 0 * BB * HH, cn + 0 * BB * HH, bar + 0, 64);

    // --- layer 1 ---
    wmma_gemm_bias_kernel<<<dim3(TT * (BB / 16), GG / 128), 256, 0, stream>>>(
        y0, wih1p, bsum1, xg, GG, HH, /*aRowStride=*/(size_t)TT * HH, /*fragC=*/1);
    lstm_rec_kernel<<<64, 256, 0, stream>>>(
        xg, whh1p, h0 + 1 * BB * HH, c0 + 1 * BB * HH,
        hbuf, y1, hn + 1 * BB * HH, cn + 1 * BB * HH, bar + 16, 64);

    // --- output projection: out[BT, 64] = y1 @ Wfc^T + bfc (row-major C) ---
    wmma_gemm_bias_kernel<<<dim3(BT / 16, 1), 128, 0, stream>>>(
        y1, wfcp, bfc, outf, OUTN, HH, /*aRowStride=*/(size_t)HH, /*fragC=*/0);
}
